// EncoderBlock_22308060135776
// MI455X (gfx1250) — compile-verified, matrix operand order checked by reading, not run on hardware
//
#include <hip/hip_runtime.h>
#include <hip/hip_bf16.h>
#include <stdint.h>
#include <stddef.h>

// ---------------- CDNA5 WMMA plumbing ----------------
typedef __attribute__((ext_vector_type(16))) __bf16          v16bf;
typedef __attribute__((ext_vector_type(8)))  float           v8f;
typedef __attribute__((ext_vector_type(8)))  unsigned short  u16x8;

struct FragU { u16x8 lo, hi; };

__device__ __forceinline__ unsigned short f2b(float f) {
  unsigned u = __builtin_bit_cast(unsigned, f);
  u += 0x7FFFu + ((u >> 16) & 1u);           // round-to-nearest-even
  return (unsigned short)(u >> 16);
}

// 16-bit A/B fragment gather (ISA 7.12.2): lane L holds row (L&15); element e maps
// to K = e + (e>=8?8:0) + (L>=16?8:0)  ->  two contiguous 16B LDS reads at
// koff = (L>>4)*8 and koff+16 within a 32-wide K tile.
__device__ __forceinline__ v16bf ld_frag(const unsigned short* rowbase, int koff) {
  FragU f;
  f.lo = *reinterpret_cast<const u16x8*>(rowbase + koff);
  f.hi = *reinterpret_cast<const u16x8*>(rowbase + koff + 16);
  return __builtin_bit_cast(v16bf, f);
}

__device__ __forceinline__ v8f wmma_bf16(v16bf a, v16bf b, v8f c) {
  return __builtin_amdgcn_wmma_f32_16x16x32_bf16(false, a, false, b, (short)0, c, false, false);
}

// ---------------- problem constants ----------------
static constexpr int CB   = 2;
static constexpr int CS   = 2048;
static constexpr int CD   = 1024;
static constexpr int CH   = 16;
static constexpr int CDFF = 4096;
static constexpr int CM   = CB * CS;   // 4096 rows

// ---------------- fp32 -> bf16 convert ----------------
__global__ __launch_bounds__(256) void f32_to_bf16_kernel(
    const float* __restrict__ in, unsigned short* __restrict__ out, int n) {
  int i = blockIdx.x * 256 + threadIdx.x;
  if (i < n) out[i] = f2b(in[i]);
}

// ---------------- W[K][N] f32 -> Wt[N][K] bf16 (one-time per weight) ----------------
__global__ __launch_bounds__(256) void w_transpose_bf16_kernel(
    const float* __restrict__ W, unsigned short* __restrict__ Wt, int K, int N) {
  __shared__ float tile[32][33];
  const int k0 = blockIdx.y * 32, n0 = blockIdx.x * 32;
  const int tx = threadIdx.x & 31, ty = threadIdx.x >> 5;   // 32 x 8
#pragma unroll
  for (int i = 0; i < 32; i += 8)
    tile[ty + i][tx] = W[(size_t)(k0 + ty + i) * N + n0 + tx];
  __syncthreads();
#pragma unroll
  for (int i = 0; i < 32; i += 8)
    Wt[(size_t)(n0 + ty + i) * K + k0 + tx] = f2b(tile[tx][ty + i]);
}

// ---------------- tiled bf16 WMMA GEMM:  C = A(bf16)[M,K] @ Bt(bf16)[N,K]^T + bias ----------
// block tile 128x128, BK=32, 4 waves (2x2), each wave 64x64 via 4x4 wmma accs.
// LDS double-buffered: global loads for tile k+1 issued before the 16 WMMAs of tile k.
__global__ __launch_bounds__(128) void gemm_bf16_kernel(
    const unsigned short* __restrict__ A, const unsigned short* __restrict__ Bt,
    const float* __restrict__ bias, float* __restrict__ outF,
    unsigned short* __restrict__ outB, int M, int N, int K, int relu) {
  __shared__ __align__(16) unsigned short As[2][128 * 32];   // [m][k]
  __shared__ __align__(16) unsigned short Bs[2][128 * 32];   // [n][k]

  const int tid  = threadIdx.x;
  const int wave = tid >> 5, lane = tid & 31;
  const int wm = wave & 1, wn = wave >> 1;
  const int lm = lane & 15, lh = lane >> 4;
  const int koff = lh * 8;
  const int bm = blockIdx.y * 128, bn = blockIdx.x * 128;

  v8f acc[4][4];
#pragma unroll
  for (int i = 0; i < 4; ++i)
#pragma unroll
    for (int j = 0; j < 4; ++j) acc[i][j] = (v8f)(0.0f);

  // stager: one 128-row x 32-k tile each of A and B; thread t owns one full row (64B).
  const unsigned short* gA = A  + (size_t)(bm + tid) * K;
  const unsigned short* gB = Bt + (size_t)(bn + tid) * K;

  {  // prologue: stage tile 0 into buffer 0
    const u16x8* pa = reinterpret_cast<const u16x8*>(gA);
    const u16x8* pb = reinterpret_cast<const u16x8*>(gB);
    u16x8* da = reinterpret_cast<u16x8*>(&As[0][tid * 32]);
    u16x8* db = reinterpret_cast<u16x8*>(&Bs[0][tid * 32]);
#pragma unroll
    for (int i = 0; i < 4; ++i) { da[i] = pa[i]; db[i] = pb[i]; }
  }
  __syncthreads();

  for (int k0 = 0; k0 < K; k0 += 32) {
    const int cur = (k0 >> 5) & 1, nxt = cur ^ 1;
    const bool have_next = (k0 + 32) < K;

    u16x8 ra[4], rb[4];
    if (have_next) {   // issue next tile's global loads before the WMMA burst
      const u16x8* pa = reinterpret_cast<const u16x8*>(gA + k0 + 32);
      const u16x8* pb = reinterpret_cast<const u16x8*>(gB + k0 + 32);
#pragma unroll
      for (int i = 0; i < 4; ++i) { ra[i] = pa[i]; rb[i] = pb[i]; }
      if (k0 + 64 < K) __builtin_prefetch(gA + k0 + 64, 0, 0);   // global_prefetch_b8
    }

    v16bf afrag[4];
#pragma unroll
    for (int ms = 0; ms < 4; ++ms)
      afrag[ms] = ld_frag(&As[cur][(wm * 64 + ms * 16 + lm) * 32], koff);
#pragma unroll
    for (int ns = 0; ns < 4; ++ns) {
      v16bf bfrag = ld_frag(&Bs[cur][(wn * 64 + ns * 16 + lm) * 32], koff);
#pragma unroll
      for (int ms = 0; ms < 4; ++ms)
        acc[ms][ns] = wmma_bf16(afrag[ms], bfrag, acc[ms][ns]);
    }

    if (have_next) {   // drain staged registers into the alternate buffer
      u16x8* da = reinterpret_cast<u16x8*>(&As[nxt][tid * 32]);
      u16x8* db = reinterpret_cast<u16x8*>(&Bs[nxt][tid * 32]);
#pragma unroll
      for (int i = 0; i < 4; ++i) { da[i] = ra[i]; db[i] = rb[i]; }
    }
    __syncthreads();
  }

#pragma unroll
  for (int ms = 0; ms < 4; ++ms)
#pragma unroll
    for (int ns = 0; ns < 4; ++ns) {
      const int col = bn + wn * 64 + ns * 16 + lm;
      const float bv = bias[col];
#pragma unroll
      for (int vv = 0; vv < 8; ++vv) {
        const int row = bm + wm * 64 + ms * 16 + lh * 8 + vv;   // C/D: m = vgpr + 8*(lane>=16)
        float val = acc[ms][ns][vv] + bv;
        if (relu) val = fmaxf(val, 0.0f);
        const size_t idx = (size_t)row * N + col;
        if (outF) outF[idx] = val;
        if (outB) outB[idx] = f2b(val);
      }
    }
}

// ---------------- flash attention: softmax(Q K^T / H + mask) V ----------------
// grid(S/64, B*H), 128 threads = 4 waves, wave = 16 query rows, key tiles of 32.
__global__ __launch_bounds__(128) void attn_kernel(
    const unsigned short* __restrict__ q, const unsigned short* __restrict__ k,
    const unsigned short* __restrict__ v, const int* __restrict__ mask,
    unsigned short* __restrict__ outb) {
  __shared__ __align__(16) unsigned short Qs[64 * 64];   // [qrow][dk]
  __shared__ __align__(16) unsigned short Ks[32 * 64];   // [key][dk]
  __shared__ __align__(16) unsigned short Vt[64 * 32];   // [dk][key] (transposed)
  __shared__ __align__(16) unsigned short Ps[4][16 * 32];
  __shared__ float Ss[4][16 * 32];
  __shared__ float mstat[4][16], lstat[4][16], cstat[4][16];
  __shared__ int msk[32];

  const int tid = threadIdx.x;
  const int wave = tid >> 5, lane = tid & 31;
  const int lm = lane & 15, lh = lane >> 4;
  const int koff = lh * 8;
  const int bh = blockIdx.y;
  const int b = bh / CH, h = bh % CH;
  const int q0 = blockIdx.x * 64;

  {  // stage Q tile
    const int r = tid >> 1, c0 = (tid & 1) * 32;
    const unsigned short* g = q + (size_t)(b * CS + q0 + r) * CD + h * 64 + c0;
#pragma unroll
    for (int i = 0; i < 4; ++i)
      *reinterpret_cast<u16x8*>(&Qs[r * 64 + c0 + i * 8]) =
          *reinterpret_cast<const u16x8*>(g + i * 8);
  }
  if (tid < 64) { mstat[tid >> 4][tid & 15] = -3.0e38f; lstat[tid >> 4][tid & 15] = 0.0f; }
  __syncthreads();

  v16bf qa[2];
#pragma unroll
  for (int ks = 0; ks < 2; ++ks)
    qa[ks] = ld_frag(&Qs[(wave * 16 + lm) * 64 + ks * 32], koff);

  v8f oacc[4];
#pragma unroll
  for (int ns = 0; ns < 4; ++ns) oacc[ns] = (v8f)(0.0f);

  const float scale = 1.0f / (float)CH;   // reference divides by num_heads

  for (int kt = 0; kt < CS / 32; ++kt) {
    {  // stage K tile + transposed V tile
      const int key = tid >> 2, c0 = (tid & 3) * 16;
      const unsigned short* gk = k + (size_t)(b * CS + kt * 32 + key) * CD + h * 64 + c0;
      *reinterpret_cast<u16x8*>(&Ks[key * 64 + c0])     = *reinterpret_cast<const u16x8*>(gk);
      *reinterpret_cast<u16x8*>(&Ks[key * 64 + c0 + 8]) = *reinterpret_cast<const u16x8*>(gk + 8);
      const unsigned short* gv = v + (size_t)(b * CS + kt * 32 + key) * CD + h * 64 + c0;
      u16x8 v0 = *reinterpret_cast<const u16x8*>(gv);
      u16x8 v1 = *reinterpret_cast<const u16x8*>(gv + 8);
#pragma unroll
      for (int i = 0; i < 8; ++i) Vt[(c0 + i) * 32 + key] = v0[i];
#pragma unroll
      for (int i = 0; i < 8; ++i) Vt[(c0 + 8 + i) * 32 + key] = v1[i];
    }
    if (tid < 32) msk[tid] = mask[b * CS + kt * 32 + tid];
    __syncthreads();

    // scores: 16x32 per wave via 4 wmma
    v8f sacc[2];
    sacc[0] = (v8f)(0.0f); sacc[1] = (v8f)(0.0f);
#pragma unroll
    for (int ns = 0; ns < 2; ++ns)
#pragma unroll
      for (int ks = 0; ks < 2; ++ks) {
        v16bf bf = ld_frag(&Ks[(ns * 16 + lm) * 64 + ks * 32], koff);
        sacc[ns] = wmma_bf16(qa[ks], bf, sacc[ns]);
      }
#pragma unroll
    for (int ns = 0; ns < 2; ++ns)
#pragma unroll
      for (int vv = 0; vv < 8; ++vv)
        Ss[wave][(lh * 8 + vv) * 32 + ns * 16 + lm] = sacc[ns][vv] * scale;
    __builtin_amdgcn_wave_barrier();

    // online softmax, one row per lane (lanes 0..15)
    if (lane < 16) {
      const int r = lane;
      float mo = mstat[wave][r];
      float mx = mo;
#pragma unroll 4
      for (int c = 0; c < 32; ++c) {
        float s = Ss[wave][r * 32 + c];
        if (msk[c] == 0) s = -1.0e9f;
        Ss[wave][r * 32 + c] = s;
        mx = fmaxf(mx, s);
      }
      const float corr = __expf(mo - mx);
      float sum = 0.0f;
#pragma unroll 4
      for (int c = 0; c < 32; ++c) {
        float p = __expf(Ss[wave][r * 32 + c] - mx);
        Ps[wave][r * 32 + c] = f2b(p);
        sum += p;
      }
      mstat[wave][r] = mx;
      lstat[wave][r] = lstat[wave][r] * corr + sum;
      cstat[wave][r] = corr;
    }
    __builtin_amdgcn_wave_barrier();

    // rescale running O, then O += P @ V
#pragma unroll
    for (int ns = 0; ns < 4; ++ns)
#pragma unroll
      for (int vv = 0; vv < 8; ++vv) oacc[ns][vv] *= cstat[wave][lh * 8 + vv];

    v16bf pa = ld_frag(&Ps[wave][lm * 32], koff);
#pragma unroll
    for (int ns = 0; ns < 4; ++ns) {
      v16bf vb = ld_frag(&Vt[(ns * 16 + lm) * 32], koff);
      oacc[ns] = wmma_bf16(pa, vb, oacc[ns]);
    }
    __syncthreads();   // before next tile overwrites Ks/Vt/msk
  }

#pragma unroll
  for (int ns = 0; ns < 4; ++ns)
#pragma unroll
    for (int vv = 0; vv < 8; ++vv) {
      const int m = lh * 8 + vv;
      const float val = oacc[ns][vv] / lstat[wave][m];
      const size_t row = (size_t)(b * CS + q0 + wave * 16 + m);
      outb[row * CD + h * 64 + ns * 16 + lm] = f2b(val);
    }
}

// ---------------- residual + LayerNorm (unbiased std, /(std+eps)) ----------------
__global__ __launch_bounds__(256) void ln_residual_kernel(
    const float* __restrict__ resid, const float* __restrict__ vin,
    const float* __restrict__ alpha, const float* __restrict__ beta,
    float* __restrict__ outF, unsigned short* __restrict__ outB, int N) {
  __shared__ float sbuf[256], qbuf[256];
  const int row = blockIdx.x, tid = threadIdx.x;
  const float* vr = vin + (size_t)row * N;
  float s = 0.0f, sq = 0.0f;
  for (int i = tid; i < N; i += 256) { float x = vr[i]; s += x; sq += x * x; }
  sbuf[tid] = s; qbuf[tid] = sq;
  __syncthreads();
  for (int st = 128; st > 0; st >>= 1) {
    if (tid < st) { sbuf[tid] += sbuf[tid + st]; qbuf[tid] += qbuf[tid + st]; }
    __syncthreads();
  }
  const float mean = sbuf[0] / (float)N;
  const float var  = fmaxf((qbuf[0] - (float)N * mean * mean) / (float)(N - 1), 0.0f);
  const float dinv = 1.0f / (sqrtf(var) + 1.0e-6f);
  for (int i = tid; i < N; i += 256) {
    const float y = (vr[i] - mean) * dinv * alpha[i] + beta[i];
    const float o = resid[(size_t)row * N + i] + y;
    outF[(size_t)row * N + i] = o;
    if (outB) outB[(size_t)row * N + i] = f2b(o);
  }
}

// ---------------- orchestration ----------------
extern "C" void kernel_launch(void* const* d_in, const int* in_sizes, int n_in,
                              void* d_out, int out_size, void* d_ws, size_t ws_size,
                              hipStream_t stream) {
  (void)in_sizes; (void)n_in; (void)out_size; (void)ws_size;
  const float* x    = (const float*)d_in[0];
  const int*   mask = (const int*)d_in[1];
  const float* Wq = (const float*)d_in[2];  const float* bq = (const float*)d_in[3];
  const float* Wk = (const float*)d_in[4];  const float* bk = (const float*)d_in[5];
  const float* Wv = (const float*)d_in[6];  const float* bv = (const float*)d_in[7];
  const float* Wo = (const float*)d_in[8];  const float* bo = (const float*)d_in[9];
  const float* W1 = (const float*)d_in[10]; const float* b1 = (const float*)d_in[11];
  const float* W2 = (const float*)d_in[12]; const float* b2 = (const float*)d_in[13];
  const float* a1 = (const float*)d_in[14]; const float* be1 = (const float*)d_in[15];
  const float* a2 = (const float*)d_in[16]; const float* be2 = (const float*)d_in[17];

  char* ws = (char*)d_ws;
  size_t off = 0;
  auto alloc = [&](size_t bytes) -> void* {
    void* p = ws + off;
    off += (bytes + 255) & ~(size_t)255;
    return p;
  };
  unsigned short* xbf   = (unsigned short*)alloc((size_t)CM * CD * 2);
  unsigned short* qb    = (unsigned short*)alloc((size_t)CM * CD * 2);
  unsigned short* kb    = (unsigned short*)alloc((size_t)CM * CD * 2);
  unsigned short* vb    = (unsigned short*)alloc((size_t)CM * CD * 2);
  unsigned short* attnb = (unsigned short*)alloc((size_t)CM * CD * 2);
  float*          aproj = (float*)alloc((size_t)CM * CD * 4);
  float*          x1    = (float*)alloc((size_t)CM * CD * 4);
  unsigned short* x1bf  = (unsigned short*)alloc((size_t)CM * CD * 2);
  unsigned short* ff1   = (unsigned short*)alloc((size_t)CM * CDFF * 2);
  float*          ff2   = (float*)alloc((size_t)CM * CD * 4);
  unsigned short* wqT = (unsigned short*)alloc((size_t)CD * CD * 2);     // [N][K] bf16
  unsigned short* wkT = (unsigned short*)alloc((size_t)CD * CD * 2);
  unsigned short* wvT = (unsigned short*)alloc((size_t)CD * CD * 2);
  unsigned short* woT = (unsigned short*)alloc((size_t)CD * CD * 2);
  unsigned short* w1T = (unsigned short*)alloc((size_t)CD * CDFF * 2);   // N=4096,K=1024
  unsigned short* w2T = (unsigned short*)alloc((size_t)CDFF * CD * 2);   // N=1024,K=4096

  // one-time weight transpose/convert (amortized over all 32 row-blocks of M)
  dim3 tDD(CD / 32, CD / 32);
  w_transpose_bf16_kernel<<<tDD, 256, 0, stream>>>(Wq, wqT, CD, CD);
  w_transpose_bf16_kernel<<<tDD, 256, 0, stream>>>(Wk, wkT, CD, CD);
  w_transpose_bf16_kernel<<<tDD, 256, 0, stream>>>(Wv, wvT, CD, CD);
  w_transpose_bf16_kernel<<<tDD, 256, 0, stream>>>(Wo, woT, CD, CD);
  dim3 t1(CDFF / 32, CD / 32);
  w_transpose_bf16_kernel<<<t1, 256, 0, stream>>>(W1, w1T, CD, CDFF);
  dim3 t2(CD / 32, CDFF / 32);
  w_transpose_bf16_kernel<<<t2, 256, 0, stream>>>(W2, w2T, CDFF, CD);

  f32_to_bf16_kernel<<<(CM * CD) / 256, 256, 0, stream>>>(x, xbf, CM * CD);

  dim3 gD(CD / 128, CM / 128);
  gemm_bf16_kernel<<<gD, 128, 0, stream>>>(xbf, wqT, bq, nullptr, qb, CM, CD, CD, 0);
  gemm_bf16_kernel<<<gD, 128, 0, stream>>>(xbf, wkT, bk, nullptr, kb, CM, CD, CD, 0);
  gemm_bf16_kernel<<<gD, 128, 0, stream>>>(xbf, wvT, bv, nullptr, vb, CM, CD, CD, 0);

  dim3 gA(CS / 64, CB * CH);
  attn_kernel<<<gA, 128, 0, stream>>>(qb, kb, vb, mask, attnb);

  gemm_bf16_kernel<<<gD, 128, 0, stream>>>(attnb, woT, bo, aproj, nullptr, CM, CD, CD, 0);
  ln_residual_kernel<<<CM, 256, 0, stream>>>(x, aproj, a1, be1, x1, x1bf, CD);

  dim3 gF(CDFF / 128, CM / 128);
  gemm_bf16_kernel<<<gF, 128, 0, stream>>>(x1bf, w1T, b1, nullptr, ff1, CM, CDFF, CD, 1);
  gemm_bf16_kernel<<<gD, 128, 0, stream>>>(ff1, w2T, b2, ff2, nullptr, CM, CD, CDFF, 0);
  ln_residual_kernel<<<CM, 256, 0, stream>>>(x1, ff2, a2, be2, (float*)d_out, nullptr, CD);
}
